// GRU4REC_678604833556
// MI455X (gfx1250) — matrix operand order repaired and checked by reading, not compiled, not run on hardware
//
#include <hip/hip_runtime.h>

#define B_SZ 1024
#define H_SZ 256
#define C_SZ 100000
#define K_SZ 256   // K == H for both GEMMs

typedef __attribute__((ext_vector_type(16))) _Float16 v16h;
typedef __attribute__((ext_vector_type(8)))  float    v8f;
typedef __attribute__((ext_vector_type(4)))  float    v4f;

union Frag16 { v16h v; uint4 q[2]; };
union Pack8  { _Float16 h[8]; uint4 q; };

// Hardware transcendental tanh (CDNA5 TANH trans op); builtin preferred, asm fallback.
__device__ inline float fast_tanh(float x) {
#if __has_builtin(__builtin_amdgcn_tanhf)
  return __builtin_amdgcn_tanhf(x);
#elif __has_builtin(__builtin_amdgcn_tanh_f32)
  return __builtin_amdgcn_tanh_f32(x);
#else
  float y;
  asm("v_tanh_f32 %0, %1" : "=v"(y) : "v"(x));
  return y;
#endif
}

// Out[M x N] = act(A[M x K] * Bmat[N x K]^T + bias[N])
//   A: row-major, K contiguous; f16 (AF16) or f32 (converted while staging)
//   Bmat: f32 row-major (N x K), K contiguous; converted to f16 into LDS
//   NT layout -> both WMMA fragments load contiguous-K chunks (no transpose loads).
//   TANH also selects non-temporal output stores (logits are write-once; keeps
//   W_out resident in the 192MB L2 across M-blocks).
template <bool AF16, bool TANH>
__global__ __launch_bounds__(256)
void gemm_nt_wmma(const void* __restrict__ Aptr,
                  const float* __restrict__ Bmat,
                  const float* __restrict__ bias,
                  float* __restrict__ Out,
                  int M, int N) {
  constexpr int K  = K_SZ;
  constexpr int BM = 64, BN = 128, KS = 32;
  constexpr int ASTR = K + 8;    // 264 halfs: frag reads bank-conflict-free
  constexpr int BSTR = KS + 8;   // 40 halfs
  constexpr int CSTR = BN + 4;   // 132 floats: conflict-free C staging
  constexpr int ABYTES = BM * ASTR * 2;              // 33792
  constexpr int SMEM_BYTES = ABYTES + 2 * BN * BSTR * 2;  // + 20480 = 54272

  __shared__ __align__(16) char smem[SMEM_BYTES];
  _Float16 (*As)[ASTR]       = (_Float16(*)[ASTR])smem;
  _Float16 (*Bs)[BN][BSTR]   = (_Float16(*)[BN][BSTR])(smem + ABYTES);
  float    (*Cs)[CSTR]       = (float(*)[CSTR])smem;   // aliases As (33792 B each)

  const int tid  = threadIdx.x;
  const int lane = tid & 31;
  const int wid  = tid >> 5;
  const int wm   = wid >> 2;   // 0..1  (M sub-block)
  const int wn   = wid & 3;    // 0..3  (N sub-block)

  const int m0 = blockIdx.y * BM;   // M == 1024: always a multiple of BM
  const int n0 = blockIdx.x * BN;

  // ---- stage A tile (covers full K) into LDS once: 8 uniform iterations ----
  {
    const int r = tid >> 5;          // 0..7 base row, +8 per iteration
    const int c = (tid & 31) * 8;    // 0..248
    if (AF16) {
      const _Float16* Ag = (const _Float16*)Aptr + (size_t)(m0 + r) * K + c;
      #pragma unroll
      for (int it = 0; it < 8; ++it)
        *(uint4*)&As[r + it * 8][c] = *(const uint4*)(Ag + (size_t)it * 8 * K);
    } else {
      const float* Ag = (const float*)Aptr + (size_t)(m0 + r) * K + c;
      #pragma unroll
      for (int it = 0; it < 8; ++it) {
        const float4 x = *(const float4*)(Ag + (size_t)it * 8 * K);
        const float4 y = *(const float4*)(Ag + (size_t)it * 8 * K + 4);
        Pack8 p;
        p.h[0] = (_Float16)x.x; p.h[1] = (_Float16)x.y;
        p.h[2] = (_Float16)x.z; p.h[3] = (_Float16)x.w;
        p.h[4] = (_Float16)y.x; p.h[5] = (_Float16)y.y;
        p.h[6] = (_Float16)y.z; p.h[7] = (_Float16)y.w;
        *(uint4*)&As[r + it * 8][c] = p.q;
      }
    }
  }

  // ---- B tile loader: 128 rows x 32 K-floats -> f16 LDS, 2 uniform chunks ----
  const int br0 = tid >> 2;            // 0..63  (chunk 0 row);  chunk 1 row = br0 + 64
  const int bcc = (tid & 3) * 8;       // 0,8,16,24
  int gn0 = n0 + br0;       gn0 = gn0 < N ? gn0 : N - 1;   // clamp keeps addr in-bounds
  int gn1 = n0 + br0 + 64;  gn1 = gn1 < N ? gn1 : N - 1;
  const float* bp0 = Bmat + (size_t)gn0 * K + bcc;
  const float* bp1 = Bmat + (size_t)gn1 * K + bcc;

  auto loadB = [&](int stage, int ks) {
    const int k0 = ks * KS;
    #pragma unroll
    for (int half = 0; half < 2; ++half) {
      const float* s = (half == 0) ? (bp0 + k0) : (bp1 + k0);
      const float4 x = *(const float4*)s;
      const float4 y = *(const float4*)(s + 4);
      Pack8 p;
      p.h[0] = (_Float16)x.x; p.h[1] = (_Float16)x.y;
      p.h[2] = (_Float16)x.z; p.h[3] = (_Float16)x.w;
      p.h[4] = (_Float16)y.x; p.h[5] = (_Float16)y.y;
      p.h[6] = (_Float16)y.z; p.h[7] = (_Float16)y.w;
      *(uint4*)&Bs[stage][br0 + half * 64][bcc] = p.q;
    }
  };

  v8f acc00 = {}, acc01 = {}, acc10 = {}, acc11 = {};

  loadB(0, 0);
  __syncthreads();

  // Fragment addressing (ISA 7.12.2, 16-bit layouts, wave32):
  //  A lane m=lane&15: K = aoff..aoff+7 then aoff+16..aoff+23, aoff = 8*(lane>=16)
  //  B lane n=lane&15: K = boff..boff+7 then boff+16..boff+23, boff = 8*(lane>=16)
  const int aoff = (lane >> 4) * 8;
  const int boff = (lane >> 4) * 8;
  const int ar   = wm * 32 + (lane & 15);
  const int brr  = wn * 32 + (lane & 15);

  #pragma unroll
  for (int ks = 0; ks < K / KS; ++ks) {
    if (ks + 1 < K / KS) loadB((ks + 1) & 1, ks + 1);   // prefetch into other buffer
    const int st = ks & 1;
    const int kb = ks * KS;

    Frag16 a0, a1, b0, b1;
    a0.q[0] = *(const uint4*)&As[ar][kb + aoff];
    a0.q[1] = *(const uint4*)&As[ar][kb + aoff + 16];
    a1.q[0] = *(const uint4*)&As[ar + 16][kb + aoff];
    a1.q[1] = *(const uint4*)&As[ar + 16][kb + aoff + 16];
    b0.q[0] = *(const uint4*)&Bs[st][brr][boff];
    b0.q[1] = *(const uint4*)&Bs[st][brr][boff + 16];
    b1.q[0] = *(const uint4*)&Bs[st][brr + 16][boff];
    b1.q[1] = *(const uint4*)&Bs[st][brr + 16][boff + 16];

    acc00 = __builtin_amdgcn_wmma_f32_16x16x32_f16(false, a0.v, false, b0.v, (short)0, acc00, false, false);
    acc01 = __builtin_amdgcn_wmma_f32_16x16x32_f16(false, a0.v, false, b1.v, (short)0, acc01, false, false);
    acc10 = __builtin_amdgcn_wmma_f32_16x16x32_f16(false, a1.v, false, b0.v, (short)0, acc10, false, false);
    acc11 = __builtin_amdgcn_wmma_f32_16x16x32_f16(false, a1.v, false, b1.v, (short)0, acc11, false, false);

    __syncthreads();   // also fences As reads before Cs aliasing overwrite below
  }

  // ---- epilogue: bias + activation -> LDS C-tile -> coalesced b128 stores ----
  // C/D layout: lane<16 -> N=lane, rows M=0..7; lane>=16 -> N=lane-16, rows M=8..15
  {
    const int ncol = lane & 15;
    const int mloc = wm * 32 + ((lane >> 4) * 8);
    #pragma unroll
    for (int sn = 0; sn < 2; ++sn) {
      const int nloc = wn * 32 + sn * 16 + ncol;
      int gb = n0 + nloc; gb = gb < N ? gb : N - 1;
      const float bv = bias[gb];
      const v8f aA = (sn == 0) ? acc00 : acc01;   // rows mloc..mloc+7
      const v8f aB = (sn == 0) ? acc10 : acc11;   // rows mloc+16..mloc+23
      #pragma unroll
      for (int r = 0; r < 8; ++r) {
        float v = aA[r] + bv;
        float w = aB[r] + bv;
        if (TANH) { v = fast_tanh(v); w = fast_tanh(w); }
        Cs[mloc + r][nloc]      = v;
        Cs[mloc + 16 + r][nloc] = w;
      }
    }
  }
  __syncthreads();

  // 8 uniform float4 copies per thread; each wave writes 512B contiguous per row.
  {
    const int r = tid >> 5;          // 0..7 base row, +8 per iteration
    const int c = (tid & 31) * 4;    // 0..124
    if (n0 + BN <= N) {              // block-uniform: full tile, unconditional
      #pragma unroll
      for (int it = 0; it < 8; ++it) {
        const v4f v = *(const v4f*)&Cs[r + it * 8][c];
        float* dst = &Out[(size_t)(m0 + r + it * 8) * N + n0 + c];
        if (TANH) __builtin_nontemporal_store(v, (v4f*)dst);
        else      *(v4f*)dst = v;
      }
    } else {                         // ragged last N-block (N % 4 == 0 -> float4-safe)
      #pragma unroll
      for (int it = 0; it < 8; ++it) {
        if (n0 + c < N) {
          const v4f v = *(const v4f*)&Cs[r + it * 8][c];
          float* dst = &Out[(size_t)(m0 + r + it * 8) * N + n0 + c];
          if (TANH) __builtin_nontemporal_store(v, (v4f*)dst);
          else      *(v4f*)dst = v;
        }
      }
    }
  }
}

// GRU gates: gi gather (one-hot matmul == column gather of W_ih) + elementwise,
// produces h_new as f16 for the big WMMA GEMM, and copies hidden to output tail.
__global__ __launch_bounds__(256)
void gru_gates(const int* __restrict__ input,
               const float* __restrict__ hidden,
               const float* __restrict__ W_ih,
               const float* __restrict__ b_ih,
               const float* __restrict__ gh,      // B x 3H, already includes b_hh
               _Float16* __restrict__ hnew16,
               float* __restrict__ hid_out) {
  const int b = blockIdx.x;
  const int t = threadIdx.x;
  const int col = input[b];

  const float i_r = W_ih[(size_t)(0 * H_SZ + t) * C_SZ + col] + b_ih[0 * H_SZ + t];
  const float i_z = W_ih[(size_t)(1 * H_SZ + t) * C_SZ + col] + b_ih[1 * H_SZ + t];
  const float i_n = W_ih[(size_t)(2 * H_SZ + t) * C_SZ + col] + b_ih[2 * H_SZ + t];

  const size_t g = (size_t)b * (3 * H_SZ);
  const float h_r = gh[g + 0 * H_SZ + t];
  const float h_z = gh[g + 1 * H_SZ + t];
  const float h_n = gh[g + 2 * H_SZ + t];

  const float r = 1.0f / (1.0f + __expf(-(i_r + h_r)));
  const float z = 1.0f / (1.0f + __expf(-(i_z + h_z)));
  const float n = fast_tanh(i_n + r * h_n);

  const float h  = hidden[(size_t)b * H_SZ + t];
  const float hn = (1.0f - z) * n + z * h;

  hnew16[b * H_SZ + t]  = (_Float16)hn;
  hid_out[b * H_SZ + t] = h;   // reference returns the *input* hidden (its bug), reproduced
}

extern "C" void kernel_launch(void* const* d_in, const int* in_sizes, int n_in,
                              void* d_out, int out_size, void* d_ws, size_t ws_size,
                              hipStream_t stream) {
  const int*   input  = (const int*)  d_in[0];
  const float* hidden = (const float*)d_in[1];   // (1, B, H)
  const float* W_ih   = (const float*)d_in[2];   // (3H, C)
  const float* W_hh   = (const float*)d_in[3];   // (3H, H)
  const float* b_ih   = (const float*)d_in[4];   // (3H,)
  const float* b_hh   = (const float*)d_in[5];   // (3H,)
  const float* W_out  = (const float*)d_in[6];   // (C, H)
  const float* b_out  = (const float*)d_in[7];   // (C,)

  float* logit   = (float*)d_out;                       // (B, C)
  float* hid_out = logit + (size_t)B_SZ * C_SZ;         // (1, B, H) tail

  // workspace: [hnew16: B*H f16][gh: B*3H f32]
  _Float16* hnew16 = (_Float16*)d_ws;
  float*    gh     = (float*)((char*)d_ws + (size_t)B_SZ * H_SZ * sizeof(_Float16));

  // 1) gh = hidden @ W_hh^T + b_hh   (M=1024, N=768, K=256; A is f32)
  gemm_nt_wmma<false, false><<<dim3(768 / 128, B_SZ / 64), 256, 0, stream>>>(
      (const void*)hidden, W_hh, b_hh, gh, B_SZ, 3 * H_SZ);

  // 2) gates -> h_new (f16) + copy hidden to output tail
  gru_gates<<<B_SZ, H_SZ, 0, stream>>>(input, hidden, W_ih, b_ih, gh, hnew16, hid_out);

  // 3) logit = tanh(h_new @ W_out^T + b_out)   (M=1024, N=100000, K=256; A is f16)
  const int nblk = (C_SZ + 127) / 128;   // 782, ragged last tile guarded
  gemm_nt_wmma<true, true><<<dim3(nblk, B_SZ / 64), 256, 0, stream>>>(
      (const void*)hnew16, W_out, b_out, logit, B_SZ, C_SZ);
}
